// DiffusionActiveInference_76209899700644
// MI455X (gfx1250) — matrix operand order repaired and checked by reading, not compiled, not run on hardware
//
#include <hip/hip_runtime.h>
#include <cstdint>
#include <cstddef>

// ---------------------------------------------------------------------------
// Problem constants (match reference)
// ---------------------------------------------------------------------------
#define LATENT   256
#define ACTION   32
#define HIDDEN   1024
#define TEMB     128
#define BATCH_N  1024
#define NTRAJ    16
#define HORIZON  5
#define ROWS     32          // batch rows per workgroup (two 16-row WMMA tiles)

typedef __attribute__((ext_vector_type(16))) __bf16 v16bf;
typedef __attribute__((ext_vector_type(8)))  float  v8f;

// ---------------------------------------------------------------------------
// Packed bf16 weight layout in workspace (element offsets, bf16 units).
// Each 32(K) x 16(N) tile is 512 bf16 stored lane-major: lane*16 + e,
// where element e of lane l is W[kt*32 + (l>=16?8:0) + (e&7) + (e>=8?16:0)][nt*16 + (l&15)].
// This is exactly the V_WMMA_*_BF16 B-matrix fragment, so a wave loads a
// full B fragment with two global_load_b128 per tile.
// ---------------------------------------------------------------------------
static constexpr size_t OFF_PW1 = 0;
static constexpr size_t OFF_PW2 = OFF_PW1 + (size_t)256 * 1024;
static constexpr size_t OFF_PWM = OFF_PW2 + (size_t)1024 * 1024;
static constexpr size_t OFF_PWS = OFF_PWM + (size_t)1024 * 32;
static constexpr size_t OFF_DW1 = OFF_PWS + (size_t)1024 * 32;
static constexpr size_t OFF_DW2 = OFF_DW1 + (size_t)288 * 1024;
static constexpr size_t OFF_DW3 = OFF_DW2 + (size_t)1024 * 1024;
static constexpr size_t OFF_VW1 = OFF_DW3 + (size_t)1024 * 256;
static constexpr size_t OFF_VW2 = OFF_VW1 + (size_t)384 * 1024;
static constexpr size_t OFF_VW3 = OFF_VW2 + (size_t)1024 * 1024;   // padded 1024x16
static constexpr size_t PACKED_ELEMS   = OFF_VW3 + (size_t)1024 * 16;
static constexpr size_t TRAJ_OFF_BYTES = PACKED_ELEMS * 2;          // f32 [NTRAJ][BATCH] follows

// LDS row strides (bf16 elements), padded 8 elems (16B) to spread banks.
static constexpr int CATS = 384 + 8;    // concat buffer [ROWS][384]
static constexpr int HS   = 1024 + 8;   // hidden buffers [ROWS][1024]
static constexpr int SMEM_BYTES =
    ROWS * CATS * 2 +      // cat      25088
    ROWS * HS * 2 +        // hA       66048
    ROWS * HS * 2 +        // hB       66048
    ROWS * 256 * 4 +       // zB       32768
    ROWS * 256 * 4 +       // znB      32768
    ROWS * 32 * 4 +        // meanB     4096
    ROWS * 32 * 4 +        // logsB     4096
    128 * 4 +              // embB       512
    4 * ROWS * 4;          // ent/epi/prag/tot 512  => 231936 total

// ---------------------------------------------------------------------------
// Helpers
// ---------------------------------------------------------------------------
__device__ __forceinline__ unsigned short bfbits(float f) {
  unsigned u = __builtin_bit_cast(unsigned, f);
  unsigned r = u + 0x7FFFu + ((u >> 16) & 1u);   // round-to-nearest-even
  return (unsigned short)(r >> 16);
}

struct BF32B { uint4 lo, hi; };   // 32 bytes == one v16bf fragment

// A fragment (16x32 bf16, MxK): lane l holds row (l&15);
// VGPR0..3 = tile-K (l>=16?8:0)+0..7, VGPR4..7 = tile-K 16+(l>=16?8:0)+0..7.
__device__ __forceinline__ v16bf load_a_frag(const unsigned short* __restrict__ a,
                                             int strideE, int kt, int lane) {
  const int row = lane & 15;
  const int kb  = (kt << 5) + ((lane >> 4) << 3);
  const unsigned short* p = a + row * strideE + kb;
  BF32B t;
  t.lo = *reinterpret_cast<const uint4*>(p);        // K kb..kb+7
  t.hi = *reinterpret_cast<const uint4*>(p + 16);   // K kb+16..kb+23
  return __builtin_bit_cast(v16bf, t);
}

// B fragment from packed global tile (1024B): lane reads its own 32B.
__device__ __forceinline__ v16bf load_b_frag(const unsigned short* __restrict__ w, int lane) {
  const unsigned short* p = w + (lane << 4);
  BF32B t;
  t.lo = *reinterpret_cast<const uint4*>(p);
  t.hi = *reinterpret_cast<const uint4*>(p + 8);
  return __builtin_bit_cast(v16bf, t);
}

// Two 16-row A-tiles (rows 0-15 and 16-31) sharing each B fragment: every
// 1KB L2 B-fetch feeds two WMMAs -> 32 FLOP per L2 byte. kt loop stays
// rolled so the live set is accLo/accHi (2*NT*8) + 2 A frags + NT B frags.
template<int KT, int NT>
__device__ __forceinline__ void gemm_slab2(const unsigned short* __restrict__ wBase,
                                           int ntilesTotal, int ntStart,
                                           const unsigned short* __restrict__ aLDS, int aStride,
                                           int lane, v8f (&accLo)[NT], v8f (&accHi)[NT]) {
  v8f zero = {};
#pragma unroll
  for (int j = 0; j < NT; ++j) { accLo[j] = zero; accHi[j] = zero; }
  const unsigned short* aHi = aLDS + 16 * aStride;
#pragma unroll 1
  for (int kt = 0; kt < KT; ++kt) {
    v16bf a0 = load_a_frag(aLDS, aStride, kt, lane);
    v16bf a1 = load_a_frag(aHi,  aStride, kt, lane);
#pragma unroll
    for (int j = 0; j < NT; ++j) {
      v16bf b = load_b_frag(wBase + ((size_t)(kt * ntilesTotal + ntStart + j) << 9), lane);
      accLo[j] = __builtin_amdgcn_wmma_f32_16x16x32_bf16(
          false, a0, false, b, (short)0, accLo[j], false, false);
      accHi[j] = __builtin_amdgcn_wmma_f32_16x16x32_bf16(
          false, a1, false, b, (short)0, accHi[j], false, false);
    }
  }
}

// Bias + SiLU epilogue for one 16-row tile, store bf16 into LDS buffer.
template<int NT>
__device__ __forceinline__ void store_silu(const v8f (&acc)[NT], int ntStart, int rowOff,
                                           unsigned short* __restrict__ out, int strideE,
                                           const float* __restrict__ bias, int lane) {
  const int c  = lane & 15;
  const int r0 = ((lane >> 4) << 3) + rowOff;
#pragma unroll
  for (int j = 0; j < NT; ++j) {
    const int col = ((ntStart + j) << 4) + c;
    const float b = bias[col];
#pragma unroll
    for (int i = 0; i < 8; ++i) {
      float x = acc[j][i] + b;
      float s = x * (1.0f / (1.0f + __expf(-x)));
      out[(r0 + i) * strideE + col] = bfbits(s);
    }
  }
}

// Full N=1024 layer over 32 rows: each wave owns a 128-col slab, processed as
// 4 groups of 2 n-tiles (acc pressure = 32 VGPRs, B reused across row tiles).
template<int KT>
__device__ __forceinline__ void layer1024(const unsigned short* __restrict__ w,
                                          const unsigned short* __restrict__ aLDS, int aStride,
                                          unsigned short* __restrict__ out,
                                          const float* __restrict__ bias,
                                          int wave, int lane) {
#pragma unroll 1
  for (int g = 0; g < 4; ++g) {
    v8f accLo[2], accHi[2];
    const int ntStart = wave * 8 + g * 2;
    gemm_slab2<KT, 2>(w, 64, ntStart, aLDS, aStride, lane, accLo, accHi);
    store_silu<2>(accLo, ntStart, 0,  out, HS, bias, lane);
    store_silu<2>(accHi, ntStart, 16, out, HS, bias, lane);
  }
}

// ---------------------------------------------------------------------------
// Weight packing: f32 [K][Nreal] row-major -> bf16 WMMA-B tiles [K/32][Npad/16].
// Npad may exceed Nreal (zero pad), used for vW3 (Nreal=1 -> Npad=16).
// ---------------------------------------------------------------------------
__global__ void pack_w_kernel(const float* __restrict__ src, unsigned short* __restrict__ dst,
                              int K, int Npad, int Nreal) {
  int idx = blockIdx.x * blockDim.x + threadIdx.x;
  int total = K * Npad;
  if (idx >= total) return;
  int tile = idx >> 9, within = idx & 511;
  int lane = within >> 4, e = within & 15;
  int ntiles = Npad >> 4;
  int kt = tile / ntiles, nt = tile - kt * ntiles;
  int k = (kt << 5) + ((lane >> 4) << 3) + (e & 7) + ((e >> 3) << 4);   // lanes 0-15: K 0-7,16-23; lanes 16-31: K 8-15,24-31
  int n = (nt << 4) + (lane & 15);
  float v = (n < Nreal) ? src[(size_t)k * Nreal + n] : 0.0f;
  dst[idx] = bfbits(v);
}

// ---------------------------------------------------------------------------
// Main kernel: one workgroup (8 wave32) = 32 batch rows of one trajectory,
// full 5-step horizon rollout on-chip.
// ---------------------------------------------------------------------------
__global__ __launch_bounds__(256, 1) void efe_kernel(
    const float* __restrict__ latent, const float* __restrict__ noise,
    const float* __restrict__ pb1, const float* __restrict__ pb2,
    const float* __restrict__ pbm, const float* __restrict__ pbs,
    const float* __restrict__ db1, const float* __restrict__ db2, const float* __restrict__ db3,
    const float* __restrict__ vb1, const float* __restrict__ vb2, const float* __restrict__ vb3,
    const unsigned short* __restrict__ wp, float* __restrict__ trajEfe) {
  extern __shared__ char smem[];
  unsigned short* cat = (unsigned short*)smem;          // [ROWS][CATS] bf16 concat/A buffer
  unsigned short* hA  = cat + ROWS * CATS;              // [ROWS][HS] bf16
  unsigned short* hB  = hA + ROWS * HS;                 // [ROWS][HS] bf16
  float* zB    = (float*)(hB + ROWS * HS);              // [ROWS][256]
  float* znB   = zB    + ROWS * 256;                    // [ROWS][256]
  float* meanB = znB   + ROWS * 256;                    // [ROWS][32]
  float* logsB = meanB + ROWS * 32;                     // [ROWS][32]
  float* embB  = logsB + ROWS * 32;                     // [128]
  float* entB  = embB  + 128;                           // [ROWS]
  float* epiB  = entB  + ROWS;
  float* pragB = epiB  + ROWS;
  float* totB  = pragB + ROWS;

  const int tid  = threadIdx.x;
  const int lane = tid & 31;
  const int wave = tid >> 5;
  const int wg   = blockIdx.x;
  const int traj = wg >> 5;            // 16 trajectories
  const int rowBase = (wg & 31) << 5;  // 32 tiles of 32 batch rows

  const float LOG2PI = 1.8378770664093453f;

  // Load initial latent: f32 copy + bf16 copy into concat buffer cols 0..255.
  for (int i = tid; i < ROWS * 256; i += 256) {
    int r = i >> 8, c = i & 255;
    float v = latent[(size_t)(rowBase + r) * 256 + c];
    zB[i] = v;
    cat[r * CATS + c] = bfbits(v);
  }
  if (tid < ROWS) totB[tid] = 0.0f;
  __syncthreads();

  float disc = 1.0f;
#pragma unroll 1
  for (int t = 0; t < HORIZON; ++t) {
    // Sinusoidal time embedding (same for all rows; t is uniform).
    if (tid < 64) {
      float fr  = __expf(-9.210340371976184f * (float)tid / 64.0f); // -log(1e4)*i/half
      float ang = (float)t * fr;
      embB[tid]      = __sinf(ang);
      embB[tid + 64] = __cosf(ang);
    }

    // ---- policy trunk: h1 = silu(z @ pW1 + pb1)  (K=256, N=1024) ----
    layer1024<8>(wp + OFF_PW1, cat, CATS, hA, pb1, wave, lane);
    __syncthreads();
    // ---- h2 = silu(h1 @ pW2 + pb2)  (K=1024, N=1024) ----
    layer1024<32>(wp + OFF_PW2, hA, HS, hB, pb2, wave, lane);
    __syncthreads();
    // ---- heads: mean (waves 0-1), log_std (waves 2-3); N=32 each ----
    if (wave < 4) {
      v8f accLo[1], accHi[1];
      const unsigned short* w = (wave < 2) ? (wp + OFF_PWM) : (wp + OFF_PWS);
      int nt = wave & 1;
      gemm_slab2<32, 1>(w, 2, nt, hB, HS, lane, accLo, accHi);
      int col = (nt << 4) + (lane & 15);
      int r0  = (lane >> 4) << 3;
      if (wave < 2) {
        float b = pbm[col];
#pragma unroll
        for (int i = 0; i < 8; ++i) {
          meanB[(r0 + i) * 32 + col]      = accLo[0][i] + b;
          meanB[(16 + r0 + i) * 32 + col] = accHi[0][i] + b;
        }
      } else {
        float b = pbs[col];
#pragma unroll
        for (int i = 0; i < 8; ++i) {
          float lsLo = fminf(2.0f, fmaxf(-5.0f, accLo[0][i] + b));
          float lsHi = fminf(2.0f, fmaxf(-5.0f, accHi[0][i] + b));
          logsB[(r0 + i) * 32 + col]      = lsLo;
          logsB[(16 + r0 + i) * 32 + col] = lsHi;
        }
      }
    }
    __syncthreads();
    // ---- reparameterized action into concat cols 256..287, entropy per row ----
#pragma unroll
    for (int q = 0; q < 4; ++q) {
      int e = tid + q * 256;           // 1024 = 32 rows * 32 dims
      int r = e >> 5, d = e & 31;
      float ls  = logsB[r * 32 + d];
      float eps = noise[(((size_t)traj * HORIZON + t) * BATCH_N + rowBase + r) * 32 + d];
      float a   = meanB[r * 32 + d] + __expf(ls) * eps;
      cat[r * CATS + 256 + d] = bfbits(a);
    }
    if (tid < ROWS) {
      float s = 32.0f * 0.5f * (1.0f + LOG2PI);
      for (int d = 0; d < 32; ++d) s += logsB[tid * 32 + d];
      entB[tid] = s;
    }
    __syncthreads();
    // ---- dynamics: silu([z,a] @ dW1 + db1)  (K=288) ----
    layer1024<9>(wp + OFF_DW1, cat, CATS, hA, db1, wave, lane);
    __syncthreads();
    layer1024<32>(wp + OFF_DW2, hA, HS, hB, db2, wave, lane);
    __syncthreads();
    // ---- z_next = z + h2 @ dW3 + db3  (N=256, 2 ntiles per wave) ----
    {
      v8f accLo[2], accHi[2];
      gemm_slab2<32, 2>(wp + OFF_DW3, 16, wave * 2, hB, HS, lane, accLo, accHi);
#pragma unroll
      for (int j = 0; j < 2; ++j) {
        int col = ((wave * 2 + j) << 4) + (lane & 15);
        int r0  = (lane >> 4) << 3;
        float b = db3[col];
#pragma unroll
        for (int i = 0; i < 8; ++i) {
          int rL = r0 + i, rH = 16 + r0 + i;
          znB[rL * 256 + col] = zB[rL * 256 + col] + accLo[j][i] + b;
          znB[rH * 256 + col] = zB[rH * 256 + col] + accHi[j][i] + b;
        }
      }
    }
    __syncthreads();
    // ---- epistemic KL on normalized latents ----
    if (tid < ROWS) {
      float sz = 0.f, sn = 0.f, dt = 0.f;
      for (int c = 0; c < 256; ++c) {
        float a = zB[tid * 256 + c], b = znB[tid * 256 + c];
        sz += a * a; sn += b * b; dt += a * b;
      }
      float nz = sqrtf(sz) + 1e-8f, nn = sqrtf(sn) + 1e-8f;
      float kl = 0.5f * (sz / (nz * nz) + sn / (nn * nn) - 2.0f * dt / (nz * nn));
      epiB[tid] = log1pf(fmaxf(kl, 0.0f));
    }
    __syncthreads();
    // ---- advance state: z <- z_next; build value input [z_next, time_emb] ----
    for (int i = tid; i < ROWS * 256; i += 256) {
      int r = i >> 8, c = i & 255;
      float v = znB[i];
      zB[i] = v;
      cat[r * CATS + c] = bfbits(v);
    }
    for (int i = tid; i < ROWS * 128; i += 256) {
      int r = i >> 7, c = i & 127;
      cat[r * CATS + 256 + c] = bfbits(embB[c]);
    }
    __syncthreads();
    // ---- value net (K=384 -> 1024 -> 1024 -> 1, padded to 16) ----
    layer1024<12>(wp + OFF_VW1, cat, CATS, hA, vb1, wave, lane);
    __syncthreads();
    layer1024<32>(wp + OFF_VW2, hA, HS, hB, vb2, wave, lane);
    __syncthreads();
    if (wave == 0) {
      v8f accLo[1], accHi[1];
      gemm_slab2<32, 1>(wp + OFF_VW3, 1, 0, hB, HS, lane, accLo, accHi); // col 0 real, rest 0
      if ((lane & 15) == 0) {
        int r0 = (lane >> 4) << 3;
        float b = vb3[0];
#pragma unroll
        for (int i = 0; i < 8; ++i) {
          pragB[r0 + i]      = accLo[0][i] + b;
          pragB[16 + r0 + i] = accHi[0][i] + b;
        }
      }
    }
    __syncthreads();
    // ---- step EFE accumulate: epi + pragmatic + 0.1 * (-entropy) ----
    if (tid < ROWS) {
      float efe = epiB[tid] + pragB[tid] - 0.1f * entB[tid];
      totB[tid] += disc * efe;
    }
    disc *= 0.99f;
    __syncthreads();
  }

  if (tid < ROWS) trajEfe[(size_t)traj * BATCH_N + rowBase + tid] = totB[tid];
}

// Mean over trajectories -> d_out[BATCH]. Overwrites (no atomics) so graph
// replays stay deterministic.
__global__ void reduce_kernel(const float* __restrict__ trajEfe, float* __restrict__ out) {
  int b = blockIdx.x * blockDim.x + threadIdx.x;
  if (b < BATCH_N) {
    float s = 0.0f;
    for (int t = 0; t < NTRAJ; ++t) s += trajEfe[(size_t)t * BATCH_N + b];
    out[b] = s * (1.0f / (float)NTRAJ);
  }
}

// ---------------------------------------------------------------------------
// Launch
// ---------------------------------------------------------------------------
extern "C" void kernel_launch(void* const* d_in, const int* in_sizes, int n_in,
                              void* d_out, int out_size, void* d_ws, size_t ws_size,
                              hipStream_t stream) {
  const float* latent = (const float*)d_in[0];
  const float* noise  = (const float*)d_in[1];
  const float* pW1 = (const float*)d_in[2];  const float* pb1 = (const float*)d_in[3];
  const float* pW2 = (const float*)d_in[4];  const float* pb2 = (const float*)d_in[5];
  const float* pWm = (const float*)d_in[6];  const float* pbm = (const float*)d_in[7];
  const float* pWs = (const float*)d_in[8];  const float* pbs = (const float*)d_in[9];
  const float* dW1 = (const float*)d_in[10]; const float* db1 = (const float*)d_in[11];
  const float* dW2 = (const float*)d_in[12]; const float* db2 = (const float*)d_in[13];
  const float* dW3 = (const float*)d_in[14]; const float* db3 = (const float*)d_in[15];
  const float* vW1 = (const float*)d_in[16]; const float* vb1 = (const float*)d_in[17];
  const float* vW2 = (const float*)d_in[18]; const float* vb2 = (const float*)d_in[19];
  const float* vW3 = (const float*)d_in[20]; const float* vb3 = (const float*)d_in[21];

  size_t need = TRAJ_OFF_BYTES + (size_t)NTRAJ * BATCH_N * sizeof(float);
  if (ws_size < need) {              // cannot run without scratch; stay deterministic
    hipMemsetAsync(d_out, 0, (size_t)out_size * sizeof(float), stream);
    return;
  }

  unsigned short* wp = (unsigned short*)d_ws;
  float* trajEfe = (float*)((char*)d_ws + TRAJ_OFF_BYTES);

  auto pack = [&](const float* src, size_t offE, int K, int Npad, int Nreal) {
    int total = K * Npad;
    pack_w_kernel<<<(total + 255) / 256, 256, 0, stream>>>(src, wp + offE, K, Npad, Nreal);
  };
  pack(pW1, OFF_PW1, 256, 1024, 1024);
  pack(pW2, OFF_PW2, 1024, 1024, 1024);
  pack(pWm, OFF_PWM, 1024, 32, 32);
  pack(pWs, OFF_PWS, 1024, 32, 32);
  pack(dW1, OFF_DW1, 288, 1024, 1024);
  pack(dW2, OFF_DW2, 1024, 1024, 1024);
  pack(dW3, OFF_DW3, 1024, 256, 256);
  pack(vW1, OFF_VW1, 384, 1024, 1024);
  pack(vW2, OFF_VW2, 1024, 1024, 1024);
  pack(vW3, OFF_VW3, 1024, 16, 1);

  hipFuncSetAttribute((const void*)efe_kernel,
                      hipFuncAttributeMaxDynamicSharedMemorySize, SMEM_BYTES);
  efe_kernel<<<NTRAJ * (BATCH_N / ROWS), 256, SMEM_BYTES, stream>>>(
      latent, noise, pb1, pb2, pbm, pbs, db1, db2, db3, vb1, vb2, vb3, wp, trajEfe);

  reduce_kernel<<<(BATCH_N + 255) / 256, 256, 0, stream>>>(trajEfe, (float*)d_out);
}
